// LearnedUpdate_53154515255689
// MI455X (gfx1250) — matrix operand order repaired and checked by reading, not compiled, not run on hardware
//
#include <hip/hip_runtime.h>
#include <hip/hip_bf16.h>
#include <math.h>

typedef unsigned int       u32;
typedef unsigned short     u16;
typedef unsigned long long u64;

typedef __attribute__((ext_vector_type(16))) __bf16 v16bf;
typedef __attribute__((ext_vector_type(8)))  float  v8f;
typedef __attribute__((ext_vector_type(4)))  unsigned int v4u;
typedef __attribute__((ext_vector_type(8)))  int    v8i;
typedef __attribute__((ext_vector_type(4)))  int    v4i;

#define DD   1024   // D
#define H1N  512
#define H2N  512
#define QQ   4
#define RHO_ 0.99f
#define BB   32768
#define MT   64     // batch rows per workgroup
#define NT   256    // threads per block (8 waves)
#define KC   32     // K chunk per WMMA pass

#if __has_builtin(__builtin_amdgcn_tensor_load_to_lds) && \
    __has_builtin(__builtin_amdgcn_s_wait_tensorcnt)
#define HAVE_TDM 1
#else
#define HAVE_TDM 0
#endif

#if __has_builtin(__builtin_amdgcn_tanhf)
#define TANHF(x) __builtin_amdgcn_tanhf(x)   // v_tanh_f32 (CDNA5 TRANS unit)
#else
#define TANHF(x) tanhf(x)
#endif

// ---- float -> bf16 (round-to-nearest-even), branchless ----
__device__ __forceinline__ u16 f2bf(float f) {
    union { float f; u32 u; } v; v.f = f;
    u32 r = v.u + 0x7FFFu + ((v.u >> 16) & 1u);
    return (u16)(r >> 16);
}

union FragAB { uint4 q[2]; v16bf v; };

// A operand = transposed weights, natural LDS layout [512 rows(n)][32 halfs(k)].
// ISA 16-bit A 16x32 layout: lane L: row = L&15; K runs {hi*8.., 16+hi*8..}
// -> two contiguous 16B runs -> two ds_load_b128, zero shuffle.
__device__ __forceinline__ v16bf read_wfrag(const u16* buf, int nt, int lane) {
    int r  = lane & 15;
    int hi = lane >> 4;
    const u16* p = buf + (nt * 16 + r) * KC + hi * 8;
    FragAB f;
    f.q[0] = *reinterpret_cast<const uint4*>(p);
    f.q[1] = *reinterpret_cast<const uint4*>(p + 16);
    return f.v;
}

// B operand = activations, natural row-major LDS [m][k].
// ISA B 32x16 layout: lane L: col m = L&15, K runs {hi*8.., 16+hi*8..}
// (matches A): two 16B runs per lane -> two ds_load_b128.
__device__ __forceinline__ v16bf read_actfrag(const u16* base, int ldHalfs,
                                              int mRow0, int k0, int lane) {
    int m  = mRow0 + (lane & 15);
    int hi = lane >> 4;
    const u16* p = base + (size_t)m * ldHalfs + k0 + hi * 8;
    FragAB f;
    f.q[0] = *reinterpret_cast<const uint4*>(p);
    f.q[1] = *reinterpret_cast<const uint4*>(p + 16);
    return f.v;
}

#if HAVE_TDM
// ---------------------------------------------------------------------------
// TDM: DMA a 2D bf16 tile (32 x rows, row stride strideElems) from global
// into LDS at lds_off (packed 64B rows). D# per CDNA5 ISA 8.3-8.6.
// 6-arg builtin: (uint32x4 g0, int32x8 g1, int32x4, int32x4, int32x8, i32 cpol)
// ---------------------------------------------------------------------------
__device__ __forceinline__ void tdm_load_wchunk(const u16* gsrc, u32 lds_off,
                                                u32 rows, u32 strideElems) {
    u64 ga = (u64)(size_t)gsrc;
    v4u g0;
    g0[0] = 1u;                                      // count=1 (valid), user mode
    g0[1] = lds_off;                                 // lds_addr (bytes)
    g0[2] = (u32)(ga & 0xFFFFFFFFu);                 // global_addr[31:0]
    g0[3] = (u32)((ga >> 32) & 0x01FFFFFFu)          // global_addr[56:32]
            | 0x80000000u;                           // type=2 ("image")
    u32 td0 = strideElems;                           // tensor_dim0 (>= tile extent)
    u32 td1 = rows;                                  // tensor_dim1
    v8i g1;
    g1[0] = 0x00010000;                              // wg_mask=0, data_size=1 (2B)
    g1[1] = (int)((td0 & 0xFFFFu) << 16);            // [63:48] = td0[15:0]
    g1[2] = (int)((td0 >> 16) | ((td1 & 0xFFFFu) << 16));
    g1[3] = (int)((td1 >> 16) | ((u32)KC << 16));    // [127:112] tile_dim0 = 32
    g1[4] = (int)(rows & 0xFFFFu);                   // tile_dim1 = rows, tile_dim2=0
    g1[5] = (int)strideElems;                        // tensor_dim0_stride[31:0]
    g1[6] = 0;                                       // stride[47:32]=0, td1_stride lo=0
    g1[7] = 0;
    v4i z4 = {0, 0, 0, 0};                           // groups 2/3 unused (2D)
    v8i z8 = {0, 0, 0, 0, 0, 0, 0, 0};
    __builtin_amdgcn_tensor_load_to_lds(g0, g1, z4, z4, z8, 0);
}
#endif

// Cooperative copy fallback: 32KB chunk, pure b128 copy.
__device__ __forceinline__ void copy_wchunk(u16* dst, const u16* gsrc,
                                            u32 rows, u32 strideElems, int tid) {
    for (u32 idx = tid; idx < rows * 4; idx += NT) {
        u32 r = idx >> 2, c = idx & 3;
        reinterpret_cast<uint4*>(dst)[idx] =
            *reinterpret_cast<const uint4*>(gsrc + (size_t)r * strideElems + c * 8);
    }
}

// 2-deep software-pipelined WMMA sweep over 16 n-tiles: loads for tile ct+1
// are issued before the WMMA of tile ct, so ds_load latency hides under the
// matrix pipe instead of an s_wait_dscnt 0 full drain per tile.
__device__ __forceinline__ void wmma_sweep(v8f* acc, const u16* cur, int ntBase,
                                           int lane, v16bf bAct) {
    v16bf aW = read_wfrag(cur, ntBase, lane);
    #pragma unroll
    for (int ct = 0; ct < 16; ++ct) {
        v16bf aWn = aW;
        if (ct < 15) aWn = read_wfrag(cur, ntBase + ct + 1, lane);
        acc[ct] = __builtin_amdgcn_wmma_f32_16x16x32_bf16(
            false, aW, false, bAct, (short)0, acc[ct], false, false);
        aW = aWn;
    }
}

// ---------------------------------------------------------------------------
// Pre-kernel: fp32 weights -> TRANSPOSED bf16 (N x K, k-contiguous) in d_ws.
// ---------------------------------------------------------------------------
__global__ void cvt_weights_t(const float* __restrict__ W1, const float* __restrict__ W2,
                              const float* __restrict__ W3,
                              u16* __restrict__ o1, u16* __restrict__ o2,
                              u16* __restrict__ o3) {
    const int n1 = H1N * 2048, n2 = H2N * H1N, n3 = DD * H2N;
    for (int i = blockIdx.x * blockDim.x + threadIdx.x; i < n1 + n2 + n3;
         i += gridDim.x * blockDim.x) {
        if (i < n1) {
            int n = i >> 11, k = i & 2047;                 // Wt1[n][k] = W1[k][n]
            o1[i] = f2bf(W1[(size_t)k * H1N + n]);
        } else if (i < n1 + n2) {
            int j = i - n1, n = j >> 9, k = j & 511;       // Wt2[n][k] = W2[k][n]
            o2[j] = f2bf(W2[(size_t)k * H2N + n]);
        } else {
            int j = i - n1 - n2, n = j >> 9, k = j & 511;  // Wt3[n][k] = W3[k][n]
            o3[j] = f2bf(W3[(size_t)k * DD + n]);
        }
    }
}

// ---------------------------------------------------------------------------
// Fused 3-layer MLP. 64 batch rows per WG; h1/h2 live in LDS; weight chunks
// DMA'd by the Tensor Data Mover into a double buffer.
// ---------------------------------------------------------------------------
__global__ __launch_bounds__(NT)
void mlp_fused(const float* __restrict__ x, const float* __restrict__ loss,
               const float* __restrict__ grad,
               const float* __restrict__ W1f, const float* __restrict__ b1,
               const float* __restrict__ b2,  const float* __restrict__ b3,
               const u16* __restrict__ Wt1, const u16* __restrict__ Wt2,
               const u16* __restrict__ Wt3,
               const float* __restrict__ alpha_raw, const int* __restrict__ tptr,
               float* __restrict__ out) {
    __shared__ __align__(16) char smem[192 * 1024];
    u16* Bw0 = reinterpret_cast<u16*>(smem);                // 32 KB weight buf 0
    u16* Bw1 = reinterpret_cast<u16*>(smem + 32 * 1024);    // 32 KB weight buf 1
    u16* H1s = reinterpret_cast<u16*>(smem + 64 * 1024);    // 64 KB h1 (64x512)
    u16* H2s = reinterpret_cast<u16*>(smem + 128 * 1024);   // 64 KB h2 (64x512)
    u16* As  = H2s;   // 4 KB layer-1 activation staging aliases H2s (dead then)

    const int tid    = threadIdx.x;
    const int lane   = tid & 31;
    const int wave   = tid >> 5;
    const int mTile  = wave & 3;           // 4 row tiles of 16 batch rows
    const int nGroup = wave >> 2;          // 2 groups of 16 n-tiles (256 feats)
    const int ntB    = nGroup * 16;
    const int mR0    = mTile << 4;
    const int row0   = (int)blockIdx.x * MT;
    const int mRow   = mR0 + (lane & 15);  // this lane's batch row (D layout: N=m)
    const int hi8    = (lane >> 4) << 3;   // D layout: VGPR v -> feature v (+8)

    // scalar factor: dot(softplus(alpha), t^[0..Q]) * rho^t
    float tf = (float)(*tptr);
    float factor = 0.f, pw = 1.f;
    #pragma unroll
    for (int i = 0; i <= QQ; ++i) { factor += log1pf(expf(alpha_raw[i])) * pw; pw *= tf; }
    factor *= powf(RHO_, tf);

    const float lossm = loss[row0 + mRow];   // rank-1 loss-column term
    v8f acc[16];

    // ================= layer 1: relu(inp @ W1 + b1) =================
    #pragma unroll
    for (int ct = 0; ct < 16; ++ct) {
        int nb = ((ntB + ct) << 4) + hi8;
        #pragma unroll
        for (int v = 0; v < 8; ++v)
            acc[ct][v] = b1[nb + v] + lossm * W1f[(size_t)2048 * H1N + nb + v];
    }
#if HAVE_TDM
    if (wave == 0) tdm_load_wchunk(Wt1, (u32)(size_t)Bw0, H1N, 2048);
#endif
    for (int i = 0; i < 64; ++i) {
        int k0 = i * KC;
        u16* cur = (i & 1) ? Bw1 : Bw0;
#if HAVE_TDM
        if (i + 1 < 64) {
            if (wave == 0) {
                tdm_load_wchunk(Wt1 + (k0 + KC), (u32)(size_t)((i & 1) ? Bw0 : Bw1),
                                H1N, 2048);
                __builtin_amdgcn_s_wait_tensorcnt(1);
            }
        } else if (wave == 0) {
            __builtin_amdgcn_s_wait_tensorcnt(0);
        }
#else
        copy_wchunk(cur, Wt1 + k0, H1N, 2048, tid);
#endif
        {   // stage activation chunk (64 rows x 32 k) fp32 -> bf16, row-major
            int r = tid >> 2, kc0 = (tid & 3) << 3, c = k0 + kc0;
            const float* s = (c < DD) ? (x    + (size_t)(row0 + r) * DD + c)
                                      : (grad + (size_t)(row0 + r) * DD + (c - DD));
            float4 f0 = reinterpret_cast<const float4*>(s)[0];
            float4 f1 = reinterpret_cast<const float4*>(s)[1];
            union { uint4 q; u32 d[4]; } o;
            o.d[0] = (u32)f2bf(f0.x) | ((u32)f2bf(f0.y) << 16);
            o.d[1] = (u32)f2bf(f0.z) | ((u32)f2bf(f0.w) << 16);
            o.d[2] = (u32)f2bf(f1.x) | ((u32)f2bf(f1.y) << 16);
            o.d[3] = (u32)f2bf(f1.z) | ((u32)f2bf(f1.w) << 16);
            reinterpret_cast<uint4*>(As)[tid] = o.q;
            if (c + KC < 2 * DD) __builtin_prefetch(s + KC, 0, 1);
        }
        __syncthreads();
        v16bf bAct = read_actfrag(As, KC, mR0, 0, lane);
        wmma_sweep(acc, cur, ntB, lane, bAct);
        __syncthreads();
    }
    #pragma unroll
    for (int ct = 0; ct < 16; ++ct) {   // ReLU -> H1s (packed b128 per tile)
        int nb = ((ntB + ct) << 4) + hi8;
        union { uint4 q; u32 d[4]; } o;
        #pragma unroll
        for (int j = 0; j < 4; ++j)
            o.d[j] = (u32)f2bf(fmaxf(acc[ct][2 * j], 0.f))
                   | ((u32)f2bf(fmaxf(acc[ct][2 * j + 1], 0.f)) << 16);
        *reinterpret_cast<uint4*>(H1s + (size_t)mRow * H1N + nb) = o.q;
    }
    __syncthreads();

    // ================= layer 2: relu(h1 @ W2 + b2) =================
    #pragma unroll
    for (int ct = 0; ct < 16; ++ct) {
        int nb = ((ntB + ct) << 4) + hi8;
        #pragma unroll
        for (int v = 0; v < 8; ++v) acc[ct][v] = b2[nb + v];
    }
#if HAVE_TDM
    if (wave == 0) tdm_load_wchunk(Wt2, (u32)(size_t)Bw0, H2N, H1N);
#endif
    for (int i = 0; i < 16; ++i) {
        int k0 = i * KC;
        u16* cur = (i & 1) ? Bw1 : Bw0;
#if HAVE_TDM
        if (i + 1 < 16) {
            if (wave == 0) {
                tdm_load_wchunk(Wt2 + (k0 + KC), (u32)(size_t)((i & 1) ? Bw0 : Bw1),
                                H2N, H1N);
                __builtin_amdgcn_s_wait_tensorcnt(1);
            }
        } else if (wave == 0) {
            __builtin_amdgcn_s_wait_tensorcnt(0);
        }
#else
        copy_wchunk(cur, Wt2 + k0, H2N, H1N, tid);
#endif
        __syncthreads();
        v16bf bAct = read_actfrag(H1s, H1N, mR0, k0, lane);
        wmma_sweep(acc, cur, ntB, lane, bAct);
        __syncthreads();
    }
    #pragma unroll
    for (int ct = 0; ct < 16; ++ct) {   // ReLU -> H2s
        int nb = ((ntB + ct) << 4) + hi8;
        union { uint4 q; u32 d[4]; } o;
        #pragma unroll
        for (int j = 0; j < 4; ++j)
            o.d[j] = (u32)f2bf(fmaxf(acc[ct][2 * j], 0.f))
                   | ((u32)f2bf(fmaxf(acc[ct][2 * j + 1], 0.f)) << 16);
        *reinterpret_cast<uint4*>(H2s + (size_t)mRow * H2N + nb) = o.q;
    }
    __syncthreads();

    // ================= layer 3: tanh(h2 @ W3 + b3) * factor =================
    for (int half = 0; half < 2; ++half) {
        const u16* Wh = Wt3 + (size_t)(half * 512) * H2N;
        #pragma unroll
        for (int ct = 0; ct < 16; ++ct) {
            int nb = half * 512 + ((ntB + ct) << 4) + hi8;
            #pragma unroll
            for (int v = 0; v < 8; ++v) acc[ct][v] = b3[nb + v];
        }
#if HAVE_TDM
        if (wave == 0) tdm_load_wchunk(Wh, (u32)(size_t)Bw0, 512, H2N);
#endif
        for (int i = 0; i < 16; ++i) {
            int k0 = i * KC;
            u16* cur = (i & 1) ? Bw1 : Bw0;
#if HAVE_TDM
            if (i + 1 < 16) {
                if (wave == 0) {
                    tdm_load_wchunk(Wh + (k0 + KC), (u32)(size_t)((i & 1) ? Bw0 : Bw1),
                                    512, H2N);
                    __builtin_amdgcn_s_wait_tensorcnt(1);
                }
            } else if (wave == 0) {
                __builtin_amdgcn_s_wait_tensorcnt(0);
            }
#else
            copy_wchunk(cur, Wh + k0, 512, H2N, tid);
#endif
            __syncthreads();
            v16bf bAct = read_actfrag(H2s, H2N, mR0, k0, lane);
            wmma_sweep(acc, cur, ntB, lane, bAct);
            __syncthreads();
        }
        #pragma unroll
        for (int ct = 0; ct < 16; ++ct) {  // tanh*factor, two b128 stores per tile
            int nb = half * 512 + ((ntB + ct) << 4) + hi8;
            float* dst = out + (size_t)(row0 + mRow) * DD + nb;
            float4 o0, o1;
            o0.x = TANHF(acc[ct][0]) * factor;  o0.y = TANHF(acc[ct][1]) * factor;
            o0.z = TANHF(acc[ct][2]) * factor;  o0.w = TANHF(acc[ct][3]) * factor;
            o1.x = TANHF(acc[ct][4]) * factor;  o1.y = TANHF(acc[ct][5]) * factor;
            o1.z = TANHF(acc[ct][6]) * factor;  o1.w = TANHF(acc[ct][7]) * factor;
            reinterpret_cast<float4*>(dst)[0] = o0;
            reinterpret_cast<float4*>(dst)[1] = o1;
        }
    }
}

// ---------------------------------------------------------------------------
extern "C" void kernel_launch(void* const* d_in, const int* in_sizes, int n_in,
                              void* d_out, int out_size, void* d_ws, size_t ws_size,
                              hipStream_t stream) {
    (void)in_sizes; (void)n_in; (void)out_size; (void)ws_size;
    const float* x         = (const float*)d_in[0];
    const float* loss      = (const float*)d_in[1];
    const float* grad      = (const float*)d_in[2];
    const float* W1        = (const float*)d_in[3];
    const float* b1        = (const float*)d_in[4];
    const float* W2        = (const float*)d_in[5];
    const float* b2        = (const float*)d_in[6];
    const float* W3        = (const float*)d_in[7];
    const float* b3        = (const float*)d_in[8];
    const float* alpha_raw = (const float*)d_in[9];
    const int*   t         = (const int*)d_in[10];

    u16* Wt1 = (u16*)d_ws;                        // 512 x 2048 bf16 (W1^T, no loss row)
    u16* Wt2 = Wt1 + (size_t)H1N * 2048;          // 512 x 512
    u16* Wt3 = Wt2 + (size_t)H2N * H1N;           // 1024 x 512

    cvt_weights_t<<<2048, NT, 0, stream>>>(W1, W2, W3, Wt1, Wt2, Wt3);
    mlp_fused<<<BB / MT, NT, 0, stream>>>(x, loss, grad, W1, b1, b2, b3,
                                          Wt1, Wt2, Wt3, alpha_raw, t, (float*)d_out);
}
